// FlashSparseAttention_47579647705795
// MI455X (gfx1250) — compile-verified, matrix operand order checked by reading, not run on hardware
//
#include <hip/hip_runtime.h>
#include <hip/hip_bf16.h>

typedef __attribute__((ext_vector_type(16))) _Float16 v16h;
typedef __attribute__((ext_vector_type(8)))  _Float16 v8h;
typedef __attribute__((ext_vector_type(8)))  float    v8f;
typedef __attribute__((ext_vector_type(4)))  int      v4i;

#define B_    2
#define S_    2048
#define H_    2048
#define NH_   16
#define NKV_  4
#define HD_   128
// GROUPS = NH_/NKV_ = 4

// ---- optional CDNA5 async global->LDS path (compile-safe guard) ------------
#if defined(__has_builtin)
#if __has_builtin(__builtin_amdgcn_global_load_async_to_lds_b128) && \
    __has_builtin(__builtin_amdgcn_s_wait_asynccnt)
#define USE_ASYNC_LDS 1
#endif
#if __has_builtin(__builtin_amdgcn_permlane16)
#define USE_PERMLANE16 1
#endif
#endif

#if defined(USE_ASYNC_LDS)
typedef __attribute__((address_space(1))) v4i* gv4i_p;  // global int4*
typedef __attribute__((address_space(3))) v4i* sv4i_p;  // LDS int4*
#define ASYNC_CP16(gsrc, ldst)                                                 \
  __builtin_amdgcn_global_load_async_to_lds_b128((gv4i_p)(gsrc),               \
                                                 (sv4i_p)(ldst), 0, 0)
#endif

// ---------------------------------------------------------------------------
// Max-reduction across each 16-lane row (lanes 0-15 and 16-31 independently).
// XOR butterflies are fixed permutations -> v_permlane16_b32 with immediate
// nibble selects (no LDS traffic); falls back to __shfl_xor (ds_bpermute).
// ---------------------------------------------------------------------------
static __device__ __forceinline__ float row16_max(float x) {
#if defined(USE_PERMLANE16)
  unsigned u;
  u = __builtin_amdgcn_permlane16(__float_as_uint(x), __float_as_uint(x),
                                  0x67452301u, 0xEFCDAB89u, false, false); // ^1
  x = fmaxf(x, __uint_as_float(u));
  u = __builtin_amdgcn_permlane16(__float_as_uint(x), __float_as_uint(x),
                                  0x54761032u, 0xDCFE98BAu, false, false); // ^2
  x = fmaxf(x, __uint_as_float(u));
  u = __builtin_amdgcn_permlane16(__float_as_uint(x), __float_as_uint(x),
                                  0x32107654u, 0xBA98FEDCu, false, false); // ^4
  x = fmaxf(x, __uint_as_float(u));
  u = __builtin_amdgcn_permlane16(__float_as_uint(x), __float_as_uint(x),
                                  0xFEDCBA98u, 0x76543210u, false, false); // ^8
  x = fmaxf(x, __uint_as_float(u));
  return x;
#else
#pragma unroll
  for (int off = 1; off < 16; off <<= 1) x = fmaxf(x, __shfl_xor(x, off, 32));
  return x;
#endif
}

// ---------------------------------------------------------------------------
// Fragment loader: 16-bit A/B matrix layout for v_wmma_f32_16x16x32_f16.
// Lane r (<16):  row/col (row0+r), K = [k0 .. k0+7] and [k0+16 .. k0+23]
// Lane 16+r:     row/col (row0+r), K = [k0+8 .. k0+15] and [k0+24 .. k0+31]
// Both chunks are contiguous 16-byte loads from row-major storage.
// ---------------------------------------------------------------------------
static __device__ __forceinline__ v16h load_frag16(const _Float16* __restrict__ base,
                                                   int ld, int row0, int k0) {
  const int lane = threadIdx.x & 31;
  const int r  = lane & 15;
  const int hl = lane >> 4;
  const _Float16* p = base + (size_t)(row0 + r) * (size_t)ld + (size_t)(k0 + hl * 8);
  v8h lo = *(const v8h*)p;
  v8h hi = *(const v8h*)(p + 16);
  return __builtin_shufflevector(lo, hi, 0,1,2,3,4,5,6,7,8,9,10,11,12,13,14,15);
}

static __device__ __forceinline__ v8f wmma_f16(v16h a, v16h b, v8f c) {
  return __builtin_amdgcn_wmma_f32_16x16x32_f16(false, a, false, b, (short)0, c,
                                                false, false);
}

// ---------------------------------------------------------------------------
// f32 -> f16 conversion
// ---------------------------------------------------------------------------
__global__ void cvt_f32_f16(const float* __restrict__ src, _Float16* __restrict__ dst,
                            int n) {
  int i = blockIdx.x * blockDim.x + threadIdx.x;
  if (i < n) dst[i] = (_Float16)src[i];
}

// ---------------------------------------------------------------------------
// GEMM: out = X[MxK] @ W[NxK]^T ; each wave computes a 32x32 tile (2x2 WMMA).
// MODE 0: store Q  as f16 [B,NH, S,HD]
// MODE 1: store K  as f16 [B,NKV,S,HD]
// MODE 2: store V  as f16 [B,NKV,HD,S]   (transposed)
// MODE 3: store f32 row-major [M,N]      (final output)
// ---------------------------------------------------------------------------
template <int MODE>
static __device__ __forceinline__ void store_tile(void* __restrict__ outp, v8f acc,
                                                  int m0, int n0, int N) {
  const int lane = threadIdx.x & 31;
  const int r  = lane & 15;
  const int hl = lane >> 4;
#pragma unroll
  for (int i = 0; i < 8; ++i) {
    const int m = m0 + i + 8 * hl;
    const int n = n0 + r;
    const float v = acc[i];
    if constexpr (MODE == 3) {
      ((float*)outp)[(size_t)m * N + n] = v;
    } else {
      const int b = m >> 11, s = m & (S_ - 1);
      const int hh = n >> 7, d = n & (HD_ - 1);
      if constexpr (MODE == 0) {
        ((_Float16*)outp)[(((size_t)b * NH_ + hh) * S_ + s) * HD_ + d] = (_Float16)v;
      } else if constexpr (MODE == 1) {
        ((_Float16*)outp)[(((size_t)b * NKV_ + hh) * S_ + s) * HD_ + d] = (_Float16)v;
      } else { // MODE 2: V transposed
        ((_Float16*)outp)[(((size_t)b * NKV_ + hh) * HD_ + d) * S_ + s] = (_Float16)v;
      }
    }
  }
}

template <int MODE>
__global__ void __launch_bounds__(256)
gemm_xwT(const _Float16* __restrict__ X, const _Float16* __restrict__ W,
         void* __restrict__ outp, int M, int N, int K) {
  const int wave = blockIdx.x * (blockDim.x >> 5) + (threadIdx.x >> 5);
  const int ntiles = N >> 5;
  const int m0 = (wave / ntiles) << 5;
  const int n0 = (wave % ntiles) << 5;
  if (m0 >= M) return;

  v8f a00 = {}, a01 = {}, a10 = {}, a11 = {};
  for (int k = 0; k < K; k += 32) {
    v16h xa0 = load_frag16(X, K, m0,      k);
    v16h xa1 = load_frag16(X, K, m0 + 16, k);
    v16h wb0 = load_frag16(W, K, n0,      k);
    v16h wb1 = load_frag16(W, K, n0 + 16, k);
    a00 = wmma_f16(xa0, wb0, a00);
    a01 = wmma_f16(xa0, wb1, a01);
    a10 = wmma_f16(xa1, wb0, a10);
    a11 = wmma_f16(xa1, wb1, a11);
  }
  store_tile<MODE>(outp, a00, m0,      n0,      N);
  store_tile<MODE>(outp, a01, m0,      n0 + 16, N);
  store_tile<MODE>(outp, a10, m0 + 16, n0,      N);
  store_tile<MODE>(outp, a11, m0 + 16, n0 + 16, N);
}

// ---------------------------------------------------------------------------
// RoPE, in f32, on f16 buffer laid out [B, nheads, S, HD].
// One thread per (b, head, s, d<64) rotation pair.
// ---------------------------------------------------------------------------
__global__ void rope_kernel(_Float16* __restrict__ q, int nheads) {
  const long long idx = (long long)blockIdx.x * blockDim.x + threadIdx.x;
  const long long total = (long long)B_ * nheads * S_ * 64;
  if (idx >= total) return;
  const int d = (int)(idx & 63);
  const int s = (int)((idx >> 6) & (S_ - 1));
  const long long hb = idx >> 17; // b*nheads + head
  const size_t base = ((size_t)hb * S_ + s) * HD_;
  const float ang = (float)s * powf(10000.0f, -(float)d * (1.0f / 64.0f));
  const float c = cosf(ang), sn = sinf(ang);
  const float x1 = (float)q[base + d];
  const float x2 = (float)q[base + d + 64];
  q[base + d]      = (_Float16)(x1 * c - x2 * sn);
  q[base + d + 64] = (_Float16)(x2 * c + x1 * sn);
}

// ---------------------------------------------------------------------------
// Causal flash attention, block-cooperative K/V staging.
// One block = 8 waves = 128 consecutive queries of one (b, head).
// Per key-block (32 keys): all 256 threads stage K (32x128) and Vt (128x32)
// into LDS via global_load_async_to_lds_b128, then each wave consumes them.
// Row max via v_permlane16 butterflies; row sum via WMMA with all-ones B.
// Q: [B,NH,S,HD] f16, K: [B,NKV,S,HD] f16, Vt: [B,NKV,HD,S] f16.
// Output: f16 row-major [B*S, NH*HD].
// ---------------------------------------------------------------------------
__global__ void __launch_bounds__(256)
flash_attn_kernel(const _Float16* __restrict__ Q, const _Float16* __restrict__ K,
                  const _Float16* __restrict__ Vt, _Float16* __restrict__ O) {
  __shared__ __align__(16) _Float16 ldsK[32 * HD_];   // 32 keys x 128 dims (8KB)
  __shared__ __align__(16) _Float16 ldsV[HD_ * 32];   // 128 dims x 32 keys (8KB)
  __shared__ __align__(16) _Float16 lds_p[8][16 * 32];// per-wave P tile (8KB)

  const int t    = threadIdx.x;
  const int wid  = t >> 5;
  const int lane = t & 31;
  const int r  = lane & 15;
  const int hl = lane >> 4;

  const int qs = blockIdx.x & (S_ / 128 - 1);      // 16 query super-tiles
  const int h  = (blockIdx.x >> 4) & (NH_ - 1);
  const int b  = blockIdx.x >> 8;
  const int q0 = qs * 128 + wid * 16;              // this wave's query tile
  const int kvh = h >> 2;                          // GROUPS = 4

  const _Float16* Qb = Q  + (size_t)(b * NH_  + h)   * S_ * HD_;
  const _Float16* Kb = K  + (size_t)(b * NKV_ + kvh) * S_ * HD_;
  const _Float16* Vb = Vt + (size_t)(b * NKV_ + kvh) * HD_ * S_;

  const float scale = 0.08838834764831845f; // 1/sqrt(128)

  // Q fragments with softmax scale pre-folded (once per wave)
  v16h qf[4];
#pragma unroll
  for (int c = 0; c < 4; ++c) {
    qf[c] = load_frag16(Qb, HD_, q0, c * 32);
#pragma unroll
    for (int tt = 0; tt < 16; ++tt)
      qf[c][tt] = (_Float16)((float)qf[c][tt] * scale);
  }

  // all-ones B fragment: P @ 1 gives per-row sums in every lane of a half
  v16h onesf;
#pragma unroll
  for (int tt = 0; tt < 16; ++tt) onesf[tt] = (_Float16)1.0f;

  v8f o[8];
#pragma unroll
  for (int c = 0; c < 8; ++c) o[c] = {};
  float m_i[8], l_i[8];
#pragma unroll
  for (int i = 0; i < 8; ++i) { m_i[i] = -1e30f; l_i[i] = 0.0f; }

  const int nkb = qs * 4 + 4;  // key blocks covering keys <= qs*128+127

  // staging addresses for this thread
  const int vrow = t >> 1;                  // 0..127 (dim)
  const int vch  = (t & 1) * 16;            // 0 / 16 (key chunk)

  for (int j = 0; j < nkb; ++j) {
    const int kb = j * 32;
    __syncthreads();  // previous iteration's consumers done before overwrite
    {
      const _Float16* srcK = Kb + (size_t)kb * HD_ + t * 16;       // contiguous tile
      _Float16*       dstK = ldsK + t * 16;
      const _Float16* srcV = Vb + (size_t)vrow * S_ + kb + vch;    // strided rows
      _Float16*       dstV = ldsV + vrow * 32 + vch;
#if defined(USE_ASYNC_LDS)
      ASYNC_CP16(srcK,     dstK);
      ASYNC_CP16(srcK + 8, dstK + 8);
      ASYNC_CP16(srcV,     dstV);
      ASYNC_CP16(srcV + 8, dstV + 8);
      __builtin_amdgcn_s_wait_asynccnt(0);
#else
      *(v8h*)(dstK)     = *(const v8h*)(srcK);
      *(v8h*)(dstK + 8) = *(const v8h*)(srcK + 8);
      *(v8h*)(dstV)     = *(const v8h*)(srcV);
      *(v8h*)(dstV + 8) = *(const v8h*)(srcV + 8);
#endif
    }
    __syncthreads();

    if (kb < q0 + 16) {  // wave-uniform causal skip; EXEC all-ones inside
      v8f slo = {}, shi = {};
#pragma unroll
      for (int c = 0; c < 4; ++c) {
        slo = wmma_f16(qf[c], load_frag16(ldsK, HD_, 0,  c * 32), slo);
        shi = wmma_f16(qf[c], load_frag16(ldsK, HD_, 16, c * 32), shi);
      }
#pragma unroll
      for (int i = 0; i < 8; ++i) {
        const int qrow = q0 + 8 * hl + i;
        float a  = slo[i];
        float bb = shi[i];
        if (kb + r > qrow)      a  = -1e30f;   // causal mask
        if (kb + 16 + r > qrow) bb = -1e30f;
        const float tm   = row16_max(fmaxf(a, bb));
        const float mold = m_i[i];
        const float mnew = fmaxf(mold, tm);
        const float pl   = __expf(a - mnew);
        const float ph   = __expf(bb - mnew);
        const float corr = __expf(mold - mnew);
        l_i[i] *= corr;
        m_i[i] = mnew;
#pragma unroll
        for (int c = 0; c < 8; ++c) o[c][i] *= corr;
        // stage P tile (16 rows x 32 keys) for C-layout -> A-layout transpose
        lds_p[wid][(8 * hl + i) * 32 + r]      = (_Float16)pl;
        lds_p[wid][(8 * hl + i) * 32 + 16 + r] = (_Float16)ph;
      }
      // reload P in WMMA A-fragment layout (same wave; DS waits auto-inserted)
      v16h pf;
      {
        const _Float16* p = &lds_p[wid][r * 32 + hl * 8];
#pragma unroll
        for (int tt = 0; tt < 8; ++tt) pf[tt] = p[tt];
#pragma unroll
        for (int tt = 0; tt < 8; ++tt) pf[8 + tt] = p[16 + tt];
      }
      // row sums of (f16-rounded) P via WMMA with all-ones B
      {
        v8f zero = {};
        v8f lsum = wmma_f16(pf, onesf, zero);
#pragma unroll
        for (int i = 0; i < 8; ++i) l_i[i] += lsum[i];
      }
#pragma unroll
      for (int c = 0; c < 8; ++c) {
        v16h vf = load_frag16(ldsV, 32, c * 16, 0); // rows = dims, contiguous keys
        o[c] = wmma_f16(pf, vf, o[c]);
      }
    }
  }

#pragma unroll
  for (int i = 0; i < 8; ++i) {
    const float inv = 1.0f / l_i[i];
#pragma unroll
    for (int c = 0; c < 8; ++c) o[c][i] *= inv;
  }
#pragma unroll
  for (int c = 0; c < 8; ++c) {
#pragma unroll
    for (int i = 0; i < 8; ++i) {
      const int row = q0 + 8 * hl + i;
      const int col = h * HD_ + c * 16 + r;
      O[(size_t)(b * S_ + row) * (NH_ * HD_) + col] = (_Float16)o[c][i];
    }
  }
}

// ---------------------------------------------------------------------------
// Host-side orchestration
// ---------------------------------------------------------------------------
extern "C" void kernel_launch(void* const* d_in, const int* in_sizes, int n_in,
                              void* d_out, int out_size, void* d_ws, size_t ws_size,
                              hipStream_t stream) {
  (void)in_sizes; (void)n_in; (void)out_size; (void)ws_size;
  const float* hs = (const float*)d_in[0];
  const float* Wq = (const float*)d_in[1];
  const float* Wk = (const float*)d_in[2];
  const float* Wv = (const float*)d_in[3];
  const float* Wo = (const float*)d_in[4];
  float* out = (float*)d_out;

  const size_t MS = (size_t)B_ * S_;        // 4096 rows
  _Float16* ws  = (_Float16*)d_ws;
  _Float16* Xh  = ws;                                   // [4096, 2048]
  _Float16* Wqh = Xh  + MS * H_;                        // [2048, 2048]
  _Float16* Wkh = Wqh + (size_t)NH_ * HD_ * H_;         // [512, 2048]
  _Float16* Wvh = Wkh + (size_t)NKV_ * HD_ * H_;        // [512, 2048]
  _Float16* Woh = Wvh + (size_t)NKV_ * HD_ * H_;        // [2048, 2048]
  _Float16* Qb  = Woh + (size_t)H_ * NH_ * HD_;         // [B,NH,S,HD]
  _Float16* Kb  = Qb  + (size_t)B_ * NH_ * S_ * HD_;    // [B,NKV,S,HD]
  _Float16* Vtb = Kb  + (size_t)B_ * NKV_ * S_ * HD_;   // [B,NKV,HD,S]
  _Float16* AOh = Vtb + (size_t)B_ * NKV_ * HD_ * S_;   // [4096, 2048]

  auto cvt = [&](const float* s, _Float16* dst, size_t n) {
    cvt_f32_f16<<<(int)((n + 255) / 256), 256, 0, stream>>>(s, dst, (int)n);
  };
  cvt(hs, Xh,  MS * H_);
  cvt(Wq, Wqh, (size_t)NH_ * HD_ * H_);
  cvt(Wk, Wkh, (size_t)NKV_ * HD_ * H_);
  cvt(Wv, Wvh, (size_t)NKV_ * HD_ * H_);
  cvt(Wo, Woh, (size_t)H_ * NH_ * HD_);

  // QKV projections (32x32 tile per wave, 8 waves/block)
  gemm_xwT<0><<<1024, 256, 0, stream>>>(Xh, Wqh, Qb,  (int)MS, NH_ * HD_,  H_);
  gemm_xwT<1><<< 256, 256, 0, stream>>>(Xh, Wkh, Kb,  (int)MS, NKV_ * HD_, H_);
  gemm_xwT<2><<< 256, 256, 0, stream>>>(Xh, Wvh, Vtb, (int)MS, NKV_ * HD_, H_);

  // RoPE on Q and K
  rope_kernel<<<(B_ * NH_  * S_ * 64) / 256, 256, 0, stream>>>(Qb, NH_);
  rope_kernel<<<(B_ * NKV_ * S_ * 64) / 256, 256, 0, stream>>>(Kb, NKV_);

  // Flash attention: 512 blocks x (8 waves = 128 queries of one head)
  flash_attn_kernel<<<512, 256, 0, stream>>>(Qb, Kb, Vtb, AOh);

  // Output projection -> f32
  gemm_xwT<3><<<1024, 256, 0, stream>>>(AOh, Woh, out, (int)MS, H_, H_);
}
